// PostNormAttentionBlock_5944234738148
// MI455X (gfx1250) — matrix operand order repaired and checked by reading, not compiled, not run on hardware
//
#include <hip/hip_runtime.h>
#include <hip/hip_fp16.h>

typedef _Float16 half_t;
typedef __attribute__((ext_vector_type(16))) _Float16 v16h;
typedef __attribute__((ext_vector_type(8)))  float    v8f;
typedef __attribute__((ext_vector_type(4)))  int      v4i;

#define EPSF 1e-5f

#if __has_builtin(__builtin_amdgcn_global_load_async_to_lds_b128) && \
    __has_builtin(__builtin_amdgcn_s_wait_asynccnt)
#define USE_ASYNC_LDS 1
#else
#define USE_ASYNC_LDS 0
#endif

#if __has_builtin(__builtin_amdgcn_update_dpp)
#define USE_DPP 1
#else
#define USE_DPP 0
#endif

// ---------------------------------------------------------------------------
// WMMA helpers (CDNA5, wave32, 16x16x32 f16 -> f32)
// ---------------------------------------------------------------------------
static __device__ __forceinline__ v8f zero8() {
  v8f z = {0.f, 0.f, 0.f, 0.f, 0.f, 0.f, 0.f, 0.f};
  return z;
}

static __device__ __forceinline__ v8f wmma_f16(v16h a, v16h b, v8f c) {
  return __builtin_amdgcn_wmma_f32_16x16x32_f16(false, a, false, b, (short)0, c,
                                                false, false);
}

// A-matrix fragment (16x32 f16), tile row-major [M][K] in LDS.
static __device__ __forceinline__ v16h load_a_frag(const half_t* tile, int ld) {
  const int lane = threadIdx.x & 31;
  const half_t* p = tile + (size_t)(lane & 15) * ld + (lane >> 4) * 8;
  union { v16h v; unsigned u[8]; } f;
  f.u[0] = *(const unsigned*)(p + 0);
  f.u[1] = *(const unsigned*)(p + 2);
  f.u[2] = *(const unsigned*)(p + 4);
  f.u[3] = *(const unsigned*)(p + 6);
  f.u[4] = *(const unsigned*)(p + 16);
  f.u[5] = *(const unsigned*)(p + 18);
  f.u[6] = *(const unsigned*)(p + 20);
  f.u[7] = *(const unsigned*)(p + 22);
  return f.v;
}

// B-matrix fragment (32x16 f16), tile stored transposed: row-major [N][K].
static __device__ __forceinline__ v16h load_b_frag(const half_t* tile, int ld) {
  const int lane = threadIdx.x & 31;
  const half_t* p = tile + (size_t)(lane & 15) * ld + (lane >> 4) * 8;
  union { v16h v; unsigned u[8]; } f;
#pragma unroll
  for (int v = 0; v < 8; ++v) f.u[v] = *(const unsigned*)(p + 2 * (v & 3) + (v >> 2) * 16);
  return f.v;
}

// ---------------------------------------------------------------------------
// Global -> LDS 16-byte staging: async direct-to-LDS when available.
// Builtin signature (from clang diagnostic): (AS1 v4i*, AS3 v4i*, imm, imm).
// ---------------------------------------------------------------------------
static __device__ __forceinline__ void copy16(half_t* dst, const half_t* src) {
#if USE_ASYNC_LDS
  __builtin_amdgcn_global_load_async_to_lds_b128(
      (__attribute__((address_space(1))) v4i*)src,
      (__attribute__((address_space(3))) v4i*)dst, 0, 0);
#else
  *(uint4*)dst = *(const uint4*)src;
#endif
}
static __device__ __forceinline__ void copy16_fence() {
#if USE_ASYNC_LDS
  __builtin_amdgcn_s_wait_asynccnt(0);
#endif
}

// ---------------------------------------------------------------------------
// Width-16 cross-lane reductions via DPP16 row ops (VALU, no DS traffic).
// ---------------------------------------------------------------------------
#if USE_DPP
template <int CTRL>
static __device__ __forceinline__ float dpp_movf(float x) {
  return __builtin_bit_cast(
      float, __builtin_amdgcn_update_dpp(0, __builtin_bit_cast(int, x), CTRL,
                                         0xF, 0xF, true));
}
#endif

static __device__ __forceinline__ float row_max16(float v) {
#if USE_DPP
  v = fmaxf(v, dpp_movf<0xB1>(v));   // quad_perm(1,0,3,2)  xor 1
  v = fmaxf(v, dpp_movf<0x4E>(v));   // quad_perm(2,3,0,1)  xor 2
  v = fmaxf(v, dpp_movf<0x141>(v));  // row_half_mirror     (8-group)
  v = fmaxf(v, dpp_movf<0x140>(v));  // row_mirror          (16-group)
#else
#pragma unroll
  for (int off = 8; off; off >>= 1) v = fmaxf(v, __shfl_xor(v, off, 16));
#endif
  return v;
}

static __device__ __forceinline__ float row_sum16(float v) {
#if USE_DPP
  v += dpp_movf<0xB1>(v);
  v += dpp_movf<0x4E>(v);
  v += dpp_movf<0x141>(v);
  v += dpp_movf<0x140>(v);
#else
#pragma unroll
  for (int off = 8; off; off >>= 1) v += __shfl_xor(v, off, 16);
#endif
  return v;
}

// ---------------------------------------------------------------------------
// fp32 -> f16 convert
// ---------------------------------------------------------------------------
__global__ void cvt_f32_f16_kernel(const float* __restrict__ in,
                                   half_t* __restrict__ out, int n) {
  int i = blockIdx.x * blockDim.x + threadIdx.x;
  if (i < n) out[i] = (half_t)in[i];
}

// ---------------------------------------------------------------------------
// GEMM: C[M][N] = A16[M][K] * W16[N][K]^T + bias, templated epilogue.
// EPI 0: store f16.  EPI 1: exact GELU, store f16.  EPI 2: +resid, store f32.
// Block tile 128x128x32, 256 threads = 8 waves, wave tile 32x64.
// ---------------------------------------------------------------------------
template <int EPI>
__global__ __launch_bounds__(256) void gemm_wmma_kernel(
    const half_t* __restrict__ A, const half_t* __restrict__ W,
    const float* __restrict__ bias, const float* __restrict__ resid,
    void* __restrict__ Cout, int M, int N, int K) {
  __shared__ half_t sA[128 * 32];
  __shared__ half_t sB[128 * 32];
  const int tid = threadIdx.x;
  const int lane = tid & 31;
  const int wv = tid >> 5;
  const int wm = wv & 3, wn = wv >> 2;
  const int m0 = blockIdx.y * 128;
  const int n0 = blockIdx.x * 128;

  v8f acc[2][4];
#pragma unroll
  for (int i = 0; i < 2; ++i)
#pragma unroll
    for (int j = 0; j < 4; ++j) acc[i][j] = zero8();

  for (int k0 = 0; k0 < K; k0 += 32) {
    // cooperative staging: A 128x32 and W 128x32, 16B chunks
#pragma unroll
    for (int it = 0; it < 2; ++it) {
      int c = tid + it * 256;
      int r = c >> 2, c8 = (c & 3) * 8;
      copy16(&sA[r * 32 + c8], A + (size_t)(m0 + r) * K + k0 + c8);
      copy16(&sB[r * 32 + c8], W + (size_t)(n0 + r) * K + k0 + c8);
    }
    if (k0 + 32 < K)
      __builtin_prefetch(A + (size_t)(m0 + (tid >> 2)) * K + k0 + 32, 0, 1);
    copy16_fence();
    __syncthreads();

    v16h a0 = load_a_frag(sA + (wm * 32 + 0) * 32, 32);
    v16h a1 = load_a_frag(sA + (wm * 32 + 16) * 32, 32);
    v16h b[4];
#pragma unroll
    for (int j = 0; j < 4; ++j)
      b[j] = load_b_frag(sB + (wn * 64 + j * 16) * 32, 32);
#pragma unroll
    for (int j = 0; j < 4; ++j) {
      acc[0][j] = wmma_f16(a0, b[j], acc[0][j]);
      acc[1][j] = wmma_f16(a1, b[j], acc[1][j]);
    }
    __syncthreads();
  }

  const int g = lane >> 4, nlo = lane & 15;
#pragma unroll
  for (int i = 0; i < 2; ++i)
#pragma unroll
    for (int j = 0; j < 4; ++j) {
      int col = n0 + wn * 64 + j * 16 + nlo;
      float bcol = bias[col];
#pragma unroll
      for (int r = 0; r < 8; ++r) {
        int row = m0 + wm * 32 + i * 16 + r + 8 * g;
        float v = acc[i][j][r] + bcol;
        size_t idx = (size_t)row * N + col;
        if (EPI == 0) {
          ((half_t*)Cout)[idx] = (half_t)v;
        } else if (EPI == 1) {
          v = 0.5f * v * (1.0f + erff(v * 0.70710678118654752f));
          ((half_t*)Cout)[idx] = (half_t)v;
        } else {
          ((float*)Cout)[idx] = v + resid[idx];
        }
      }
    }
}

// ---------------------------------------------------------------------------
// Flash attention: grid (S/128, H, B), 256 threads = 8 waves x 16 q-rows.
// hd = 64, key tile = 32, online softmax, WMMA for QK^T and P*V.
// ---------------------------------------------------------------------------
__global__ __launch_bounds__(256) void attention_kernel(
    const half_t* __restrict__ q16, const half_t* __restrict__ k16,
    const half_t* __restrict__ v16p, half_t* __restrict__ o16, int B_, int S_,
    int D_, int H_) {
  const int hd = D_ / H_;  // 64
  __shared__ half_t sQ[128 * 64];
  __shared__ half_t sK[32 * 64];      // [key][d]
  __shared__ half_t sVt[64 * 32];     // [d][key] (transposed)
  __shared__ half_t sP[8 * 16 * 32];  // per-wave P tiles
  const int tid = threadIdx.x, lane = tid & 31, wv = tid >> 5;
  const int h = blockIdx.y, b = blockIdx.z;
  const int s0r = blockIdx.x * 128;
  const size_t baseH = (size_t)b * S_ * D_ + (size_t)h * hd;

  // stage Q tile (128 x 64)
  for (int c = tid; c < 1024; c += 256) {
    int r = c >> 3, d8 = (c & 7) * 8;
    copy16(&sQ[r * 64 + d8], q16 + baseH + (size_t)(s0r + r) * D_ + d8);
  }
  copy16_fence();
  __syncthreads();
  v16h qf0 = load_a_frag(sQ + wv * 16 * 64 + 0, 64);   // d 0..31
  v16h qf1 = load_a_frag(sQ + wv * 16 * 64 + 32, 64);  // d 32..63

  v8f o[4];
  float m[8], l[8];
#pragma unroll
  for (int r = 0; r < 8; ++r) { m[r] = -1e30f; l[r] = 0.f; }
#pragma unroll
  for (int j = 0; j < 4; ++j) o[j] = zero8();

  half_t* sPw = sP + wv * 16 * 32;
  const int g = lane >> 4, nlo = lane & 15;
  const float scale = 0.125f;  // 1/sqrt(64)

  for (int kt = 0; kt < S_; kt += 32) {
    // cooperative K tile (async) + transposed V tile (register transpose)
    {
      int r = tid >> 3, d8 = (tid & 7) * 8;
      const size_t gofs = baseH + (size_t)(kt + r) * D_ + d8;
      copy16(&sK[r * 64 + d8], k16 + gofs);
      uint4 uv = *(const uint4*)(v16p + gofs);
      const half_t* hh = (const half_t*)&uv;
#pragma unroll
      for (int e = 0; e < 8; ++e) sVt[(d8 + e) * 32 + r] = hh[e];
    }
    copy16_fence();
    __syncthreads();

    // scores S = Q K^T (two 16x16 fragments covering 32 keys)
    v8f s0 = zero8(), s1 = zero8();
    s0 = wmma_f16(qf0, load_b_frag(sK + 0 * 64 + 0, 64), s0);
    s0 = wmma_f16(qf1, load_b_frag(sK + 0 * 64 + 32, 64), s0);
    s1 = wmma_f16(qf0, load_b_frag(sK + 16 * 64 + 0, 64), s1);
    s1 = wmma_f16(qf1, load_b_frag(sK + 16 * 64 + 32, 64), s1);

    // online softmax (row stats live per (r, lane-group)); DPP reductions
#pragma unroll
    for (int r = 0; r < 8; ++r) {
      float x0 = s0[r] * scale, x1 = s1[r] * scale;
      float rm = row_max16(fmaxf(x0, x1));
      float mn = fmaxf(m[r], rm);
      float al = __expf(m[r] - mn);
      m[r] = mn;
      float p0 = __expf(x0 - mn), p1 = __expf(x1 - mn);
      float rs = row_sum16(p0 + p1);
      l[r] = l[r] * al + rs;
#pragma unroll
      for (int j = 0; j < 4; ++j) o[j][r] *= al;
      int prow = r + 8 * g;
      sPw[prow * 32 + nlo] = (half_t)p0;
      sPw[prow * 32 + 16 + nlo] = (half_t)p1;
    }
    __syncthreads();

    // O += P * V  (K = 32 keys, one WMMA K-step; 4 d-tiles)
    v16h pa = load_a_frag(sPw, 32);
#pragma unroll
    for (int j = 0; j < 4; ++j)
      o[j] = wmma_f16(pa, load_b_frag(sVt + j * 16 * 32, 32), o[j]);
    __syncthreads();
  }

  // normalize and write merged-head layout [B,S,D]
#pragma unroll
  for (int j = 0; j < 4; ++j)
#pragma unroll
    for (int r = 0; r < 8; ++r) {
      float vo = o[j][r] / l[r];
      int row = s0r + wv * 16 + r + 8 * g;
      o16[baseH + (size_t)row * D_ + j * 16 + nlo] = (half_t)vo;
    }
}

// ---------------------------------------------------------------------------
// LayerNorm over D=1024, one block (256 thr) per row; f32 out + optional f16.
// ---------------------------------------------------------------------------
__global__ __launch_bounds__(256) void layernorm_kernel(
    const float* __restrict__ in, const float* __restrict__ wgt,
    const float* __restrict__ bia, float* __restrict__ out,
    half_t* __restrict__ out16, int D_) {
  const int row = blockIdx.x;
  const int tid = threadIdx.x, lane = tid & 31, wv = tid >> 5;
  __shared__ float red[2][8];
  float vals[4];
  float s = 0.f, ss = 0.f;
#pragma unroll
  for (int i = 0; i < 4; ++i) {
    float v = in[(size_t)row * D_ + tid + i * 256];
    vals[i] = v;
    s += v;
    ss += v * v;
  }
#pragma unroll
  for (int off = 16; off; off >>= 1) {
    s += __shfl_xor(s, off, 32);
    ss += __shfl_xor(ss, off, 32);
  }
  if (lane == 0) { red[0][wv] = s; red[1][wv] = ss; }
  __syncthreads();
  if (tid == 0) {
    float S = 0.f, SS = 0.f;
    for (int j = 0; j < 8; ++j) { S += red[0][j]; SS += red[1][j]; }
    red[0][0] = S;
    red[1][0] = SS;
  }
  __syncthreads();
  float mu = red[0][0] / (float)D_;
  float var = red[1][0] / (float)D_ - mu * mu;
  float rstd = rsqrtf(var + EPSF);
#pragma unroll
  for (int i = 0; i < 4; ++i) {
    int col = tid + i * 256;
    float v = (vals[i] - mu) * rstd * wgt[col] + bia[col];
    out[(size_t)row * D_ + col] = v;
    if (out16) out16[(size_t)row * D_ + col] = (half_t)v;
  }
}

// ---------------------------------------------------------------------------
// Host-side orchestration
// ---------------------------------------------------------------------------
extern "C" void kernel_launch(void* const* d_in, const int* in_sizes, int n_in,
                              void* d_out, int out_size, void* d_ws,
                              size_t ws_size, hipStream_t stream) {
  const int B = 2, S = 2048, D = 1024, H = 16, F = 4096;
  const int M = B * S;

  const float* x    = (const float*)d_in[0];
  const float* Wq   = (const float*)d_in[1];
  const float* bq   = (const float*)d_in[2];
  const float* Wk   = (const float*)d_in[3];
  const float* bk   = (const float*)d_in[4];
  const float* Wv   = (const float*)d_in[5];
  const float* bv   = (const float*)d_in[6];
  const float* Wy   = (const float*)d_in[7];
  const float* by   = (const float*)d_in[8];
  const float* W1   = (const float*)d_in[9];
  const float* b1   = (const float*)d_in[10];
  const float* W2   = (const float*)d_in[11];
  const float* b2   = (const float*)d_in[12];
  const float* ln1w = (const float*)d_in[13];
  const float* ln1b = (const float*)d_in[14];
  const float* ln2w = (const float*)d_in[15];
  const float* ln2b = (const float*)d_in[16];
  (void)in_sizes; (void)n_in; (void)out_size; (void)ws_size;

  char* wsp = (char*)d_ws;
  auto carve = [&](size_t bytes) -> void* {
    void* p = (void*)wsp;
    wsp += (bytes + 255) & ~(size_t)255;
    return p;
  };
  half_t* x16   = (half_t*)carve((size_t)M * D * 2);
  half_t* wq16  = (half_t*)carve((size_t)D * D * 2);
  half_t* wk16  = (half_t*)carve((size_t)D * D * 2);
  half_t* wv16  = (half_t*)carve((size_t)D * D * 2);
  half_t* wy16  = (half_t*)carve((size_t)D * D * 2);
  half_t* w116  = (half_t*)carve((size_t)F * D * 2);
  half_t* w216  = (half_t*)carve((size_t)D * F * 2);
  half_t* q16   = (half_t*)carve((size_t)M * D * 2);
  half_t* k16   = (half_t*)carve((size_t)M * D * 2);
  half_t* v16b  = (half_t*)carve((size_t)M * D * 2);
  half_t* o16   = (half_t*)carve((size_t)M * D * 2);
  half_t* h16   = (half_t*)carve((size_t)M * F * 2);
  half_t* x1_16 = (half_t*)carve((size_t)M * D * 2);
  float*  y1    = (float*)carve((size_t)M * D * 4);
  float*  x1    = (float*)carve((size_t)M * D * 4);
  float*  y2    = (float*)carve((size_t)M * D * 4);

  auto cvt = [&](const float* src, half_t* dst, int n) {
    cvt_f32_f16_kernel<<<(n + 255) / 256, 256, 0, stream>>>(src, dst, n);
  };
  cvt(x, x16, M * D);
  cvt(Wq, wq16, D * D);
  cvt(Wk, wk16, D * D);
  cvt(Wv, wv16, D * D);
  cvt(Wy, wy16, D * D);
  cvt(W1, w116, F * D);
  cvt(W2, w216, D * F);

  dim3 blk(256);
  // QKV projections
  gemm_wmma_kernel<0><<<dim3(D / 128, M / 128), blk, 0, stream>>>(
      x16, wq16, bq, nullptr, q16, M, D, D);
  gemm_wmma_kernel<0><<<dim3(D / 128, M / 128), blk, 0, stream>>>(
      x16, wk16, bk, nullptr, k16, M, D, D);
  gemm_wmma_kernel<0><<<dim3(D / 128, M / 128), blk, 0, stream>>>(
      x16, wv16, bv, nullptr, v16b, M, D, D);
  // flash attention
  attention_kernel<<<dim3(S / 128, H, B), blk, 0, stream>>>(q16, k16, v16b,
                                                            o16, B, S, D, H);
  // output projection + residual
  gemm_wmma_kernel<2><<<dim3(D / 128, M / 128), blk, 0, stream>>>(
      o16, wy16, by, x, y1, M, D, D);
  layernorm_kernel<<<M, blk, 0, stream>>>(y1, ln1w, ln1b, x1, x1_16, D);
  // MLP
  gemm_wmma_kernel<1><<<dim3(F / 128, M / 128), blk, 0, stream>>>(
      x1_16, w116, b1, nullptr, h16, M, F, D);
  gemm_wmma_kernel<2><<<dim3(D / 128, M / 128), blk, 0, stream>>>(
      h16, w216, b2, x1, y2, M, D, F);
  layernorm_kernel<<<M, blk, 0, stream>>>(y2, ln2w, ln2b, (float*)d_out,
                                          nullptr, D);
}
